// MLPVPhi_90254442758429
// MI455X (gfx1250) — compile-verified, matrix operand order checked by reading, not run on hardware
//
#include <hip/hip_runtime.h>
#include <math.h>

typedef __attribute__((ext_vector_type(16))) _Float16 v16h;
typedef __attribute__((ext_vector_type(8)))  float    v8f;

#define Bsz 4
#define Tsz 512
#define Ssz 512
#define Dsz 128
#define HIDN 64

// ---------------------------------------------------------------------------
// Branchless exact-gelu: 0.5*v*(1+erf(v/sqrt(2))), erf via Abramowitz-Stegun
// 7.1.26 (|err| <= 1.5e-7), mapped to v_rcp_f32 + v_exp_f32 + 6 FMAs.
// No exec-mask branches (unlike ocml erff).
// ---------------------------------------------------------------------------
__device__ __forceinline__ float gelu_exact(float v) {
    const float x  = v * 0.70710678118654752f;     // v / sqrt(2)
    const float ax = __builtin_fabsf(x);
    const float t  = __builtin_amdgcn_rcpf(__builtin_fmaf(0.3275911f, ax, 1.0f));
    float p = __builtin_fmaf(t, 1.061405429f, -1.453152027f);
    p = __builtin_fmaf(t, p, 1.421413741f);
    p = __builtin_fmaf(t, p, -0.284496736f);
    p = __builtin_fmaf(t, p, 0.254829592f);
    p = p * t;
    // e^(-x^2) = exp2(-x^2 * log2(e))  -> single v_exp_f32
    const float e = __builtin_amdgcn_exp2f(ax * ax * -1.4426950408889634f);
    float r = __builtin_fmaf(-p, e, 1.0f);         // erf(|x|)
    r = __builtin_copysignf(r, x);                 // erf(x)
    return 0.5f * v * (1.0f + r);
}

// ---------------------------------------------------------------------------
// Kernel 1: pq[bt][h] = h[bt]·(W1q+W1d) + b1[h] ; pk[bs][h] = h_src[bs]·(W1k−W1d)
// ---------------------------------------------------------------------------
__global__ void prep_proj(const float* __restrict__ h,
                          const float* __restrict__ h_src,
                          const float* __restrict__ W1,
                          const float* __restrict__ b1,
                          float* __restrict__ pq,
                          float* __restrict__ pk) {
    int id = blockIdx.x * blockDim.x + threadIdx.x;
    const int NPQ = Bsz * Tsz * HIDN;
    if (id >= 2 * NPQ) return;
    int sel = (id >= NPQ) ? 1 : 0;
    int lid = id - sel * NPQ;
    int r   = lid / HIDN;
    int hc  = lid - r * HIDN;
    const float* src = sel ? h_src : h;
    const float* wc  = W1 + hc;          // column hc, row stride = HIDN
    float acc = sel ? 0.0f : b1[hc];
    #pragma unroll 4
    for (int d = 0; d < Dsz; ++d) {
        float w;
        if (sel) w = wc[(Dsz + d) * HIDN] - wc[(2 * Dsz + d) * HIDN];   // W1k - W1d
        else     w = wc[d * HIDN]         + wc[(2 * Dsz + d) * HIDN];   // W1q + W1d
        acc += src[r * Dsz + d] * w;
    }
    if (sel) pk[lid] = acc; else pq[lid] = acc;
}

// ---------------------------------------------------------------------------
// Kernel 2: pre-swizzle W2 (64x64 f32, row major) into f16 WMMA B-fragment
// layout.  Index: (( ks*4 + nt )*32 + lane)*16 + e, per ISA 7.12.2
// (16-bit operand 16x32):
//   half = lane>>4 ;  K = ks*32 + 16*(e>=8) + 8*half + (e&7) ;  N = (lane&15)+16*nt
// ---------------------------------------------------------------------------
__global__ void pack_w2(const float* __restrict__ W2, _Float16* __restrict__ w2h) {
    int id = blockIdx.x * blockDim.x + threadIdx.x;   // 2*4*32*16 = 4096
    if (id >= 4096) return;
    int e    = id & 15;
    int lane = (id >> 4) & 31;
    int nt   = (id >> 9) & 3;
    int ks   = id >> 11;
    int half = lane >> 4;
    int K = ks * 32 + ((e >= 8) ? 16 : 0) + half * 8 + (e & 7);
    int N = (lane & 15) + 16 * nt;
    w2h[id] = (_Float16)W2[K * HIDN + N];
}

// ---------------------------------------------------------------------------
// Kernel 3: fused pairwise MLP.  One wave = 16 s-values for a fixed (b,t).
// A fragment built on the fly from gelu(pq + pk); 8x v_wmma_f32_16x16x32_f16
// accumulate the full 16x64 row@W2; epilogue does gelu(+b2), ·W3, cross-lane
// reduction, +b3.
// ---------------------------------------------------------------------------
__global__ void __launch_bounds__(256)
fused_mlp(const float* __restrict__ pq,
          const float* __restrict__ pk,
          const _Float16* __restrict__ w2h,
          const float* __restrict__ b2,
          const float* __restrict__ W3,
          const float* __restrict__ b3,
          float* __restrict__ out) {
    const int lane = threadIdx.x & 31;
    const int wave = threadIdx.x >> 5;
    const int half = lane >> 4;
    const int lrow = lane & 15;
    const int b    = blockIdx.z;
    const int t    = blockIdx.y;
    const int s_base = blockIdx.x * 128 + wave * 16;

    const float* pqr = pq + (b * Tsz + t) * HIDN;               // broadcast row
    const float* pkr = pk + (b * Ssz + s_base + lrow) * HIDN;   // this lane's M-row

    v8f acc0 = {}, acc1 = {}, acc2 = {}, acc3 = {};

    const v16h* wbase = (const v16h*)w2h;
    #pragma unroll
    for (int ks = 0; ks < 2; ++ks) {
        const int kb = ks * 32 + half * 8;
        v16h a;
        #pragma unroll
        for (int e = 0; e < 8; ++e) {
            float v = pqr[kb + e] + pkr[kb + e];
            a[e] = (_Float16)gelu_exact(v);
        }
        #pragma unroll
        for (int e = 0; e < 8; ++e) {
            float v = pqr[kb + 16 + e] + pkr[kb + 16 + e];
            a[8 + e] = (_Float16)gelu_exact(v);
        }
        const v16h* wb = wbase + ks * 128 + lane;   // (ks*4+nt)*32 + lane
        acc0 = __builtin_amdgcn_wmma_f32_16x16x32_f16(false, a, false, wb[0 * 32], (short)0, acc0, false, false);
        acc1 = __builtin_amdgcn_wmma_f32_16x16x32_f16(false, a, false, wb[1 * 32], (short)0, acc1, false, false);
        acc2 = __builtin_amdgcn_wmma_f32_16x16x32_f16(false, a, false, wb[2 * 32], (short)0, acc2, false, false);
        acc3 = __builtin_amdgcn_wmma_f32_16x16x32_f16(false, a, false, wb[3 * 32], (short)0, acc3, false, false);
    }

    // Epilogue: y = gelu(acc + b2), scalar = y · W3, reduce over N (16 lanes/half)
    const float w30 = W3[lrow], w31 = W3[lrow + 16], w32 = W3[lrow + 32], w33 = W3[lrow + 48];
    const float b20 = b2[lrow], b21 = b2[lrow + 16], b22 = b2[lrow + 32], b23 = b2[lrow + 48];

    float part[8];
    #pragma unroll
    for (int i = 0; i < 8; ++i) {
        part[i] = gelu_exact(acc0[i] + b20) * w30
                + gelu_exact(acc1[i] + b21) * w31
                + gelu_exact(acc2[i] + b22) * w32
                + gelu_exact(acc3[i] + b23) * w33;
    }
    // xor-reduce across the 16 lanes of each half-wave (stays within half)
    #pragma unroll
    for (int off = 1; off <= 8; off <<= 1) {
        #pragma unroll
        for (int i = 0; i < 8; ++i)
            part[i] += __shfl_xor(part[i], off, 32);
    }

    if (lrow < 8) {
        // lane j (in half h) writes row M = j + 8*h  (C/D layout, ISA 7.12.2)
        int m = lrow + 8 * half;
        int s = s_base + m;
        out[(size_t)(b * Tsz + t) * Ssz + s] = part[lrow] + b3[0];
    }
}

// ---------------------------------------------------------------------------
extern "C" void kernel_launch(void* const* d_in, const int* in_sizes, int n_in,
                              void* d_out, int out_size, void* d_ws, size_t ws_size,
                              hipStream_t stream) {
    (void)in_sizes; (void)n_in; (void)out_size; (void)ws_size;
    const float* h     = (const float*)d_in[0];
    const float* h_src = (const float*)d_in[1];
    const float* W1    = (const float*)d_in[2];
    const float* b1    = (const float*)d_in[3];
    const float* W2    = (const float*)d_in[4];
    const float* b2    = (const float*)d_in[5];
    const float* W3    = (const float*)d_in[6];
    const float* b3    = (const float*)d_in[7];
    float* out = (float*)d_out;

    char* ws = (char*)d_ws;
    float*    pq  = (float*)ws;                     // B*T*64 f32 = 512 KB
    float*    pk  = (float*)(ws + 524288);          // B*S*64 f32 = 512 KB
    _Float16* w2h = (_Float16*)(ws + 1048576);      // 4096 f16 = 8 KB

    const int nproj = 2 * Bsz * Tsz * HIDN;
    prep_proj<<<(nproj + 255) / 256, 256, 0, stream>>>(h, h_src, W1, b1, pq, pk);
    pack_w2<<<16, 256, 0, stream>>>(W2, w2h);

    dim3 grid(Ssz / 128, Tsz, Bsz);   // 4 x 512 x 4 = 8192 blocks, 8 waves each
    fused_mlp<<<grid, 256, 0, stream>>>(pq, pk, w2h, b2, W3, b3, out);
}
